// ContextTransformer_27204322853315
// MI455X (gfx1250) — matrix operand order repaired and checked by reading, not compiled, not run on hardware
//
#include <hip/hip_runtime.h>
#include <hip/hip_bf16.h>

// ---------------------------------------------------------------------------
// MI455X / gfx1250 implementation.
// Compute-bound transformer: route all matmul FLOPs through
// v_wmma_f32_16x16x32_f16 (fp16 A/B, fp32 accum). Weights are converted and
// transposed to f16 [N][K] on the fly so B-fragments are contiguous LDS reads
// matching the wave32 WMMA B layout. Attention is a fused flash-style kernel
// with the relative-position bias computed per k-tile via a band GEMM against
// the rel embedding and a shifted scatter-add into the score tile in LDS.
// ---------------------------------------------------------------------------

typedef _Float16 h16;
typedef __attribute__((ext_vector_type(16))) _Float16 v16h;
typedef __attribute__((ext_vector_type(8)))  _Float16 v8h;
typedef __attribute__((ext_vector_type(8)))  float    v8f;

__device__ __forceinline__ v8f wmma16(v16h a, v16h b, v8f c) {
  return __builtin_amdgcn_wmma_f32_16x16x32_f16(
      /*neg_a=*/false, a, /*neg_b=*/false, b,
      /*c_mod=*/(short)0, c, /*reuse_a=*/false, /*reuse_b=*/false);
}

// A-matrix fragment (16x32 f16) from row-major LDS [rows][ld] halves.
// Contiguous k order used here (same as B): lanes 0-15 -> M=lane, K=0..15;
//                lanes 16-31 -> M=lane-16, K=16..31.
__device__ __forceinline__ v16h load_afrag(const h16* p, int ld, int mbase, int lane) {
  const int r  = mbase + (lane & 15);
  const int k0 = (lane & 16) ? 16 : 0;
  return *(const v16h*)(p + r * ld + k0);
}

// B-matrix fragment (32x16 f16) from LDS stored [N][ld] halves (K-major rows).
// Contiguous k order used here, same as A: lanes 0-15 -> N=lane, K=0..15; lanes 16-31 -> N=lane-16, K=16..31.
__device__ __forceinline__ v16h load_bfrag(const h16* p, int ld, int nbase, int lane) {
  const int r  = nbase + (lane & 15);
  const int k0 = (lane & 16) ? 16 : 0;
  return *(const v16h*)(p + r * ld + k0);
}

// ---------------------------------------------------------------------------
// Generic WMMA GEMM:  out = epilogue(A[M,K] * W[K,N])
//   A  : f16 row-major [M][K]
//   BT : f16 [N][K] (weight pre-transposed so both global & LDS reads are
//        contiguous and B-fragments come straight out of LDS)
//   epilogue: +bias, activation (0 none / 1 PReLU / 2 ReLU), +residual(f32),
//   stores f32 and/or f16.
// Block: 256 threads = 8 waves (2x4), tile 128x128, K-step 32.
// ---------------------------------------------------------------------------
__global__ __launch_bounds__(256) void gemm_wmma_kernel(
    const h16* __restrict__ A, const h16* __restrict__ BT,
    const float* __restrict__ bias, const float* __restrict__ slope,
    const float* __restrict__ residual,
    float* __restrict__ outF, h16* __restrict__ outH,
    int M, int N, int K, int act)
{
  __shared__ __align__(64) h16 As[128 * 32];
  __shared__ __align__(64) h16 Bs[128 * 32];
  const int tid = threadIdx.x, lane = tid & 31, wave = tid >> 5;
  const int wm = wave >> 2;          // 0..1 : 64 rows each
  const int wn = wave & 3;           // 0..3 : 32 cols each
  const int bm = blockIdx.y * 128, bn = blockIdx.x * 128;

  v8f acc[4][2];
  {
    const v8f zf = {};
#pragma unroll
    for (int i = 0; i < 4; ++i) { acc[i][0] = zf; acc[i][1] = zf; }
  }

  for (int k0 = 0; k0 < K; k0 += 32) {
#pragma unroll
    for (int i = 0; i < 2; ++i) {              // 512 16-byte chunks / 256 thr
      int c = tid + 256 * i;
      int r = c >> 2, kk = (c & 3) * 8;
      *(v8h*)(As + r * 32 + kk) = *(const v8h*)(A  + (size_t)(bm + r) * K + k0 + kk);
      *(v8h*)(Bs + r * 32 + kk) = *(const v8h*)(BT + (size_t)(bn + r) * K + k0 + kk);
    }
    if (k0 + 32 < K) {                          // global_prefetch_b8 next tile
      int r = tid >> 2, kk = (tid & 3) * 8;
      __builtin_prefetch(A  + (size_t)(bm + r) * K + k0 + 32 + kk, 0, 1);
      __builtin_prefetch(BT + (size_t)(bn + r) * K + k0 + 32 + kk, 0, 1);
    }
    __syncthreads();
    v16h bf0 = load_bfrag(Bs, 32, wn * 32, lane);
    v16h bf1 = load_bfrag(Bs, 32, wn * 32 + 16, lane);
#pragma unroll
    for (int i = 0; i < 4; ++i) {
      v16h af = load_afrag(As, 32, wm * 64 + i * 16, lane);
      acc[i][0] = wmma16(af, bf0, acc[i][0]);
      acc[i][1] = wmma16(af, bf1, acc[i][1]);
    }
    __syncthreads();
  }

  const float sl = slope ? *slope : 0.f;
  const int coli = lane & 15, rofs = (lane >> 4) * 8;
#pragma unroll
  for (int i = 0; i < 4; ++i)
#pragma unroll
    for (int j = 0; j < 2; ++j) {
      const int col = bn + wn * 32 + j * 16 + coli;
      const float bv = bias ? bias[col] : 0.f;
#pragma unroll
      for (int g = 0; g < 8; ++g) {
        const int row = bm + wm * 64 + i * 16 + rofs + g;
        float vv = acc[i][j][g] + bv;
        if (act == 1)      vv = (vv >= 0.f) ? vv : sl * vv;
        else if (act == 2) vv = fmaxf(vv, 0.f);
        if (residual) vv += residual[(size_t)row * N + col];
        if (outF) outF[(size_t)row * N + col] = vv;
        if (outH) outH[(size_t)row * N + col] = (h16)vv;
      }
    }
}

// ---------------------------------------------------------------------------
// Fused flash attention with relative-position bias (one block per (b,h,q64)).
// Per 128-key tile: ac = Q K^T (WMMA) -> LDS f32 scores; bd band GEMM against
// rel rows (WMMA) scatter-added with the rel-shift; online softmax (stats in
// LDS, p written f16 in-place); O += P V (WMMA, V staged transposed in LDS).
// ---------------------------------------------------------------------------
__global__ __launch_bounds__(256) void attention_rel_kernel(
    const h16* __restrict__ q, const h16* __restrict__ k,
    const h16* __restrict__ v, const h16* __restrict__ rel,   // [1024][64] f16
    h16* __restrict__ out)
{
  constexpr int S = 512, H = 16, DH = 64;
  __shared__ __align__(64) h16  qs[64 * 64];     //  8 KB  Q tile [m][d]
  __shared__ __align__(64) h16  kvs[128 * 64];   // 16 KB  K rows / rel band / V^T
  __shared__ __align__(64) float sc[64 * 128];   // 32 KB  score tile (p16 in-place)
  __shared__ float m_s[64], l_s[64], fac[64];

  const int tid = threadIdx.x, lane = tid & 31, wave = tid >> 5;
  const int q0 = blockIdx.x * 64, hh = blockIdx.y, b = blockIdx.z;
  const size_t rstride = (size_t)H * DH;         // 1024 halves
  const h16* qb = q + ((size_t)b * S) * rstride + hh * DH;
  const h16* kb = k + ((size_t)b * S) * rstride + hh * DH;
  const h16* vb = v + ((size_t)b * S) * rstride + hh * DH;

#pragma unroll
  for (int i = 0; i < 2; ++i) {                  // load Q tile 64x64
    int c = tid + 256 * i;
    int r = c >> 3, d = (c & 7) * 8;
    *(v8h*)(qs + r * 64 + d) = *(const v8h*)(qb + (size_t)(q0 + r) * rstride + d);
  }
  if (tid < 64) { m_s[tid] = -3.0e38f; l_s[tid] = 0.f; }

  const int coli = lane & 15, rofs = (lane >> 4) * 8;
  const float scale = 0.125f;                    // 1/sqrt(dh=64)
  const int wmO = wave >> 2, wnO = wave & 3;     // P*V: 32 rows x 16 cols/wave
  v8f oacc[2];
  { const v8f zf = {}; oacc[0] = zf; oacc[1] = zf; }

  for (int kt = 0; kt < S; kt += 128) {
    __syncthreads();                             // guard kvs/sc reuse + Q ready
    // ---- load K tile [128][64] (B layout: N=key row, K=dh) ----
#pragma unroll
    for (int i = 0; i < 4; ++i) {
      int c = tid + 256 * i;
      int r = c >> 3, d = (c & 7) * 8;
      *(v8h*)(kvs + r * 64 + d) = *(const v8h*)(kb + (size_t)(kt + r) * rstride + d);
    }
    __syncthreads();
    // ---- ac = Q K^T : 64x128 (waves 2x4), K=64 -> 2 wmma steps ----
    {
      const int wm = wave >> 2, wn = wave & 3;
      v8f acc[2][2];
      { const v8f zf = {}; acc[0][0]=zf; acc[0][1]=zf; acc[1][0]=zf; acc[1][1]=zf; }
#pragma unroll
      for (int ks = 0; ks < 64; ks += 32) {
        v16h bf0 = load_bfrag(kvs + ks, 64, wn * 32, lane);
        v16h bf1 = load_bfrag(kvs + ks, 64, wn * 32 + 16, lane);
#pragma unroll
        for (int i = 0; i < 2; ++i) {
          v16h af = load_afrag(qs + ks, 64, wm * 32 + i * 16, lane);
          acc[i][0] = wmma16(af, bf0, acc[i][0]);
          acc[i][1] = wmma16(af, bf1, acc[i][1]);
        }
      }
#pragma unroll
      for (int i = 0; i < 2; ++i)
#pragma unroll
        for (int j = 0; j < 2; ++j)
#pragma unroll
          for (int g = 0; g < 8; ++g) {
            int row = wm * 32 + i * 16 + rofs + g;
            int col = wn * 32 + j * 16 + coli;
            sc[row * 128 + col] = acc[i][j][g];
          }
    }
    __syncthreads();
    // ---- bd: rel-bias band, rows rb0..rb0+191 of rel, 3 tiles of 64 ----
    const int rb0 = kt - q0 + 448;               // multiple of 64, in [0,1023]
    for (int nb = 0; nb < 3; ++nb) {
#pragma unroll
      for (int i = 0; i < 2; ++i) {
        int c = tid + 256 * i;
        int r = c >> 3, d = (c & 7) * 8;
        *(v8h*)(kvs + r * 64 + d) =
            *(const v8h*)(rel + (size_t)(rb0 + nb * 64 + r) * 64 + d);
      }
      __syncthreads();
      const int wm2 = wave >> 1, wn2 = wave & 1; // 16 rows x 32 cols / wave
      v8f acc[2];
      { const v8f zf = {}; acc[0] = zf; acc[1] = zf; }
#pragma unroll
      for (int ks = 0; ks < 64; ks += 32) {
        v16h af = load_afrag(qs + ks, 64, wm2 * 16, lane);
        acc[0] = wmma16(af, load_bfrag(kvs + ks, 64, wn2 * 32, lane), acc[0]);
        acc[1] = wmma16(af, load_bfrag(kvs + ks, 64, wn2 * 32 + 16, lane), acc[1]);
      }
      // rel-shift scatter-add: band col c maps to key kl = c + row - 63
#pragma unroll
      for (int j = 0; j < 2; ++j) {
        int c = nb * 64 + wn2 * 32 + j * 16 + coli;
#pragma unroll
        for (int g = 0; g < 8; ++g) {
          int row = wm2 * 16 + rofs + g;
          int kl = c + row - 63;
          if (kl >= 0 && kl < 128) sc[row * 128 + kl] += acc[j][g];
        }
      }
      __syncthreads();
    }
    // ---- online softmax on this 128-wide tile; p -> f16 in-place ----
    for (int rr = 0; rr < 8; ++rr) {
      int row = wave * 8 + rr;
      float vv[4]; float mx = -3.0e38f;
#pragma unroll
      for (int j = 0; j < 4; ++j) {
        float t = sc[row * 128 + lane + 32 * j] * scale;
        vv[j] = t; mx = fmaxf(mx, t);
      }
#pragma unroll
      for (int o = 16; o > 0; o >>= 1) mx = fmaxf(mx, __shfl_xor(mx, o, 32));
      float mold = m_s[row], mnew = fmaxf(mold, mx);
      float sum = 0.f;
#pragma unroll
      for (int j = 0; j < 4; ++j) { float e = __expf(vv[j] - mnew); vv[j] = e; sum += e; }
#pragma unroll
      for (int o = 16; o > 0; o >>= 1) sum += __shfl_xor(sum, o, 32);
      if (lane == 0) {
        float f = __expf(mold - mnew);
        m_s[row] = mnew; l_s[row] = l_s[row] * f + sum; fac[row] = f;
      }
      h16* prow = (h16*)(sc + row * 128);
#pragma unroll
      for (int j = 0; j < 4; ++j) prow[lane + 32 * j] = (h16)vv[j];
    }
    __syncthreads();
    // ---- rescale O and stage V transposed [d][128] ----
#pragma unroll
    for (int i = 0; i < 2; ++i)
#pragma unroll
      for (int g = 0; g < 8; ++g) {
        int row = wmO * 32 + i * 16 + rofs + g;
        oacc[i][g] *= fac[row];
      }
#pragma unroll
    for (int i = 0; i < 4; ++i) {
      int c = tid + 256 * i;
      int r = c >> 3, d = (c & 7) * 8;
      v8h vvv = *(const v8h*)(vb + (size_t)(kt + r) * rstride + d);
#pragma unroll
      for (int e = 0; e < 8; ++e) kvs[(d + e) * 128 + r] = vvv[e];
    }
    __syncthreads();
    // ---- O += P V (A = p16 rows with stride 256 halves, K=128) ----
#pragma unroll
    for (int ks = 0; ks < 128; ks += 32) {
      v16h bf = load_bfrag(kvs + ks, 128, wnO * 16, lane);
#pragma unroll
      for (int i = 0; i < 2; ++i) {
        v16h af = load_afrag((const h16*)sc + ks, 256, wmO * 32 + i * 16, lane);
        oacc[i] = wmma16(af, bf, oacc[i]);
      }
    }
  }
  // ---- final normalize and store f16 ----
  h16* ob = out + ((size_t)b * S) * rstride + hh * DH;
#pragma unroll
  for (int i = 0; i < 2; ++i)
#pragma unroll
    for (int g = 0; g < 8; ++g) {
      int row = wmO * 32 + i * 16 + rofs + g;
      int d = wnO * 16 + coli;
      ob[(size_t)(q0 + row) * rstride + d] = (h16)(oacc[i][g] / l_s[row]);
    }
}

// ---------------------------------------------------------------------------
// Support kernels
// ---------------------------------------------------------------------------
__global__ void cvt_f32_to_f16(const float* __restrict__ in, h16* __restrict__ out, int n) {
  int i = blockIdx.x * 256 + threadIdx.x;
  if (i < n) out[i] = (h16)in[i];
}

// f32 [K][N] -> f16 [N][K]
__global__ void cvt_transpose_f16(const float* __restrict__ in, h16* __restrict__ out,
                                  int K, int N) {
  int i = blockIdx.x * 256 + threadIdx.x;
  if (i < K * N) {
    int kk = i / N, nn = i - kk * N;
    out[(size_t)nn * K + kk] = (h16)in[i];
  }
}

__global__ __launch_bounds__(256) void layernorm_to_f16(
    const float* __restrict__ x, const float* __restrict__ gamma,
    const float* __restrict__ beta, h16* __restrict__ out, int D)
{
  const int row = blockIdx.x;
  const float* xr = x + (size_t)row * D;
  const int tid = threadIdx.x, lane = tid & 31, wave = tid >> 5;
  float s = 0.f, s2 = 0.f;
  for (int i = tid; i < D; i += 256) { float t = xr[i]; s += t; s2 += t * t; }
#pragma unroll
  for (int o = 16; o > 0; o >>= 1) { s += __shfl_xor(s, o, 32); s2 += __shfl_xor(s2, o, 32); }
  __shared__ float rs[8], rs2[8];
  if (lane == 0) { rs[wave] = s; rs2[wave] = s2; }
  __syncthreads();
  if (wave == 0) {
    s  = (lane < 8) ? rs[lane]  : 0.f;
    s2 = (lane < 8) ? rs2[lane] : 0.f;
#pragma unroll
    for (int o = 4; o > 0; o >>= 1) { s += __shfl_xor(s, o, 32); s2 += __shfl_xor(s2, o, 32); }
    if (lane == 0) {
      float mean = s / D;
      float var  = s2 / D - mean * mean;
      rs[0] = mean; rs2[0] = __frsqrt_rn(var + 1e-5f);
    }
  }
  __syncthreads();
  const float mean = rs[0], rstd = rs2[0];
  for (int i = tid; i < D; i += 256)
    out[(size_t)row * D + i] = (h16)((xr[i] - mean) * rstd * gamma[i] + beta[i]);
}

// keyframe MLP layer 1 (K=2, trivial VALU): out16 = prelu(kp @ w1 + b1)
__global__ void kf_mlp1_kernel(const float* __restrict__ kp, const float* __restrict__ w1,
                               const float* __restrict__ b1, const float* __restrict__ a,
                               h16* __restrict__ out, int M, int N) {
  int i = blockIdx.x * 256 + threadIdx.x;
  if (i >= M * N) return;
  int m = i / N, n = i - m * N;
  float t = kp[m * 2] * w1[n] + kp[m * 2 + 1] * w1[N + n] + b1[n];
  float sl = *a;
  out[i] = (h16)(t >= 0.f ? t : sl * t);
}

// rel-position MLP: 1023 rows of rel (+ zero pad row 1023) -> f16 [1024][64]
__global__ __launch_bounds__(256) void rel_mlp_kernel(
    const float* __restrict__ w1, const float* __restrict__ b1,
    const float* __restrict__ a,  const float* __restrict__ w2,
    const float* __restrict__ b2, h16* __restrict__ out)
{
  int r = blockIdx.x * 256 + threadIdx.x;
  if (r >= 1024) return;
  if (r == 1023) { for (int d = 0; d < 64; ++d) out[r * 64 + d] = (h16)0.f; return; }
  const float p  = (float)(r - 511);
  const float sl = *a;
  float h1[64];
#pragma unroll
  for (int j = 0; j < 64; ++j) {
    float t = p * w1[j] + b1[j];
    h1[j] = t >= 0.f ? t : sl * t;
  }
  for (int d = 0; d < 64; ++d) {
    float s = b2[d];
    for (int j = 0; j < 64; ++j) s += h1[j] * w2[j * 64 + d];
    out[r * 64 + d] = (h16)s;
  }
}

// ---------------------------------------------------------------------------
extern "C" void kernel_launch(void* const* d_in, const int* in_sizes, int n_in,
                              void* d_out, int out_size, void* d_ws, size_t ws_size,
                              hipStream_t stream) {
  (void)in_sizes; (void)n_in; (void)out_size; (void)ws_size;
  constexpr int M = 4096;                     // B*S
  const float* x       = (const float*)d_in[0];
  const float* kp      = (const float*)d_in[1];
  const float* enc_w1  = (const float*)d_in[3];
  const float* enc_b1  = (const float*)d_in[4];
  const float* enc_a1  = (const float*)d_in[5];
  const float* enc_w2  = (const float*)d_in[6];
  const float* enc_b2  = (const float*)d_in[7];
  const float* enc_a2  = (const float*)d_in[8];
  const float* kf_w1   = (const float*)d_in[9];
  const float* kf_b1   = (const float*)d_in[10];
  const float* kf_a    = (const float*)d_in[11];
  const float* kf_w2   = (const float*)d_in[12];
  const float* kf_b2   = (const float*)d_in[13];
  const float* rel_w1  = (const float*)d_in[14];
  const float* rel_b1  = (const float*)d_in[15];
  const float* rel_a   = (const float*)d_in[16];
  const float* rel_w2  = (const float*)d_in[17];
  const float* rel_b2  = (const float*)d_in[18];
  const float* att_wq  = (const float*)d_in[19];
  const float* att_wk  = (const float*)d_in[20];
  const float* att_wv  = (const float*)d_in[21];
  const float* att_wo  = (const float*)d_in[22];
  const float* att_g   = (const float*)d_in[23];
  const float* att_b   = (const float*)d_in[24];
  const float* pff_w1  = (const float*)d_in[25];
  const float* pff_b1  = (const float*)d_in[26];
  const float* pff_w2  = (const float*)d_in[27];
  const float* pff_b2  = (const float*)d_in[28];
  const float* pff_g   = (const float*)d_in[29];
  const float* pff_b   = (const float*)d_in[30];
  const float* fin_g   = (const float*)d_in[31];
  const float* fin_b   = (const float*)d_in[32];
  const float* dec_w1  = (const float*)d_in[33];
  const float* dec_b1  = (const float*)d_in[34];
  const float* dec_a   = (const float*)d_in[35];
  const float* dec_w2  = (const float*)d_in[36];
  const float* dec_b2  = (const float*)d_in[37];

  // workspace layout (~66 MB)
  char* ws   = (char*)d_ws;
  float* h   = (float*)ws;                                     // 16 MB f32 residual
  h16* hn16  = (h16*)(ws + (size_t)16 * 1024 * 1024);          //  8 MB LN out
  h16* bufA  = (h16*)(ws + (size_t)24 * 1024 * 1024);          // 32 MB activations
  h16* x16   = (h16*)(ws + (size_t)56 * 1024 * 1024);          //  1 MB
  h16* wbuf  = (h16*)(ws + (size_t)57 * 1024 * 1024);          //  8 MB weight ring
  h16* rel16 = (h16*)(ws + (size_t)65 * 1024 * 1024);          // 128 KB
  h16* q16   = bufA;
  h16* k16   = bufA + (size_t)4 * 1024 * 1024;
  h16* v16   = bufA + (size_t)8 * 1024 * 1024;
  h16* att16 = bufA + (size_t)12 * 1024 * 1024;

  auto gemm = [&](const h16* Ain, const float* W, int K, int N,
                  const float* bias, const float* slope, int act,
                  const float* res, float* oF, h16* oH) {
    int total = K * N;
    cvt_transpose_f16<<<(total + 255) / 256, 256, 0, stream>>>(W, wbuf, K, N);
    gemm_wmma_kernel<<<dim3(N / 128, M / 128), dim3(256), 0, stream>>>(
        Ain, wbuf, bias, slope, res, oF, oH, M, N, K, act);
  };

  // encoder MLP
  cvt_f32_to_f16<<<(M * 128 + 255) / 256, 256, 0, stream>>>(x, x16, M * 128);
  gemm(x16,  enc_w1, 128, 512,  enc_b1, enc_a1, 1, nullptr, nullptr, bufA);
  gemm(bufA, enc_w2, 512, 1024, enc_b2, enc_a2, 1, nullptr, h, nullptr);
  // keyframe positional embedding: h += mlp(kp)
  kf_mlp1_kernel<<<(M * 1024 + 255) / 256, 256, 0, stream>>>(kp, kf_w1, kf_b1, kf_a,
                                                             bufA, M, 1024);
  gemm(bufA, kf_w2, 1024, 1024, kf_b2, nullptr, 0, h, h, nullptr);
  // relative position embedding table (f16, padded to 1024 rows)
  rel_mlp_kernel<<<4, 256, 0, stream>>>(rel_w1, rel_b1, rel_a, rel_w2, rel_b2, rel16);

  for (int i = 0; i < 6; ++i) {
    const float* wq = att_wq + (size_t)i * 1024 * 1024;
    const float* wk = att_wk + (size_t)i * 1024 * 1024;
    const float* wv = att_wv + (size_t)i * 1024 * 1024;
    const float* wo = att_wo + (size_t)i * 1024 * 1024;
    // pre-LN + QKV
    layernorm_to_f16<<<M, 256, 0, stream>>>(h, att_g + i * 1024, att_b + i * 1024,
                                            hn16, 1024);
    gemm(hn16, wq, 1024, 1024, nullptr, nullptr, 0, nullptr, nullptr, q16);
    gemm(hn16, wk, 1024, 1024, nullptr, nullptr, 0, nullptr, nullptr, k16);
    gemm(hn16, wv, 1024, 1024, nullptr, nullptr, 0, nullptr, nullptr, v16);
    // fused relative attention
    attention_rel_kernel<<<dim3(8, 16, 8), dim3(256), 0, stream>>>(q16, k16, v16,
                                                                   rel16, att16);
    // output projection + residual
    gemm(att16, wo, 1024, 1024, nullptr, nullptr, 0, h, h, nullptr);
    // pre-LN FFN
    layernorm_to_f16<<<M, 256, 0, stream>>>(h, pff_g + i * 1024, pff_b + i * 1024,
                                            hn16, 1024);
    gemm(hn16, pff_w1 + (size_t)i * 1024 * 4096, 1024, 4096,
         pff_b1 + (size_t)i * 4096, nullptr, 2, nullptr, nullptr, bufA);
    gemm(bufA, pff_w2 + (size_t)i * 4096 * 1024, 4096, 1024,
         pff_b2 + (size_t)i * 1024, nullptr, 0, h, h, nullptr);
  }

  // final LN + decoder MLP
  layernorm_to_f16<<<M, 256, 0, stream>>>(h, fin_g, fin_b, hn16, 1024);
  gemm(hn16, dec_w1, 1024, 512, dec_b1, dec_a, 1, nullptr, nullptr, bufA);
  gemm(bufA, dec_w2, 512, 128, dec_b2, nullptr, 0, nullptr, (float*)d_out, nullptr);
}
